// BiDirectionalGRUBlock_75204877353112
// MI455X (gfx1250) — matrix operand order repaired and checked by reading, not compiled, not run on hardware
//
#include <hip/hip_runtime.h>
#include <hip/hip_bf16.h>
#include <math.h>

typedef __attribute__((ext_vector_type(16))) __bf16 v16bf;
typedef __attribute__((ext_vector_type(8)))  float  v8f;

union ABf { v16bf v; unsigned int u[8]; };
union CFr { v8f v; float f[8]; };
union BfPun { __bf16 h; unsigned short u; };

#define B_DIM 32
#define T_DIM 1024
#define I_DIM 512
#define H_DIM 512
#define G3    1536
#define M_ROWS (B_DIM * T_DIM)   // 32768

// LDS strides (in dwords): rows 16B-aligned AND bank-conflict-free gathers
#define SA 20    // 80B  = 5*16;  20*m mod 64 distinct for m=0..15
#define SW 260   // 1040B = 65*16; 4*m mod 64 distinct for m=0..15

// Async copy: 16 bytes per lane, global -> LDS, tracked by ASYNCcnt.
__device__ __forceinline__ void async_ld16(unsigned ldsOff, const void* g) {
  asm volatile("global_load_async_to_lds_b128 %0, %1, off"
               :: "v"(ldsOff), "v"(g) : "memory");
}
__device__ __forceinline__ void wait_async0() {
  asm volatile("s_wait_asynccnt 0x0" ::: "memory");
}

// ---------------------------------------------------------------------------
// 1) LayerNorm: x[B,T,I] fp32 -> xn[T,B,I] bf16 (row = t*32+b feeds the GEMMs)
// ---------------------------------------------------------------------------
__global__ __launch_bounds__(256) void ln_kernel(const float* __restrict__ x,
                                                 const float* __restrict__ gamma,
                                                 const float* __restrict__ beta,
                                                 __bf16* __restrict__ xn) {
  int row = blockIdx.x;                 // row = b*1024 + t
  int bb = row >> 10, tt = row & 1023;
  const float* xr = x + (size_t)row * I_DIM;
  int tid = threadIdx.x;
  float v0 = xr[tid], v1 = xr[tid + 256];
  float s = v0 + v1, q = v0 * v0 + v1 * v1;
  #pragma unroll
  for (int o = 16; o > 0; o >>= 1) { s += __shfl_xor(s, o); q += __shfl_xor(q, o); }
  __shared__ float ps[8], pq[8], stats[2];
  int wv = tid >> 5, ln = tid & 31;
  if (ln == 0) { ps[wv] = s; pq[wv] = q; }
  __syncthreads();
  if (tid == 0) {
    float S = 0.f, Q = 0.f;
    #pragma unroll
    for (int i = 0; i < 8; ++i) { S += ps[i]; Q += pq[i]; }
    float mu = S * (1.0f / 512.0f);
    float var = Q * (1.0f / 512.0f) - mu * mu;
    stats[0] = mu; stats[1] = rsqrtf(var + 1e-5f);
  }
  __syncthreads();
  float mu = stats[0], rs = stats[1];
  __bf16* dst = xn + ((size_t)tt * 32 + bb) * I_DIM;
  dst[tid]       = (__bf16)((v0 - mu) * rs * gamma[tid]       + beta[tid]);
  dst[tid + 256] = (__bf16)((v1 - mu) * rs * gamma[tid + 256] + beta[tid + 256]);
}

// ---------------------------------------------------------------------------
// 2) Weight fp32 -> bf16 (both dirs): wih16[dir][1536][512], whh16 same
// ---------------------------------------------------------------------------
__global__ __launch_bounds__(256) void cvt_kernel(const float* __restrict__ wihf,
                                                  const float* __restrict__ wihb,
                                                  const float* __restrict__ whhf,
                                                  const float* __restrict__ whhb,
                                                  __bf16* __restrict__ wih,
                                                  __bf16* __restrict__ whh) {
  size_t idx = (size_t)blockIdx.x * blockDim.x + threadIdx.x;
  if (idx >= (size_t)G3 * I_DIM) return;
  wih[idx]                      = (__bf16)wihf[idx];
  wih[(size_t)G3 * I_DIM + idx] = (__bf16)wihb[idx];
  whh[idx]                      = (__bf16)whhf[idx];
  whh[(size_t)G3 * I_DIM + idx] = (__bf16)whhb[idx];
}

// ---------------------------------------------------------------------------
// 3) Input projection GEMM: xp[dir][32768][1536] = xn @ wih[dir]^T + b_ih
//    64x64 WG tile, 4 waves, bf16 WMMA 16x16x32.
//    Double-buffered LDS fed by ASYNC global->LDS b128 copies: tile kt+1
//    streams in while WMMAs for tile kt execute.
// ---------------------------------------------------------------------------
__global__ __launch_bounds__(128) void gemm_xp(const __bf16* __restrict__ xn,
                                               const __bf16* __restrict__ wih,
                                               const float* __restrict__ bihf,
                                               const float* __restrict__ bihb,
                                               float* __restrict__ xp) {
  const int dir   = blockIdx.z;
  const int nBase = blockIdx.x * 64;
  const int mBase = blockIdx.y * 64;
  const uint4* xnQ = (const uint4*)xn;                                   // 8 bf16 / quad
  const uint4* wQ  = (const uint4*)wih + (size_t)dir * ((size_t)G3 * I_DIM / 8);
  float* xpD = xp + (size_t)dir * (size_t)M_ROWS * G3;
  const float* bih = dir ? bihb : bihf;

  __shared__ __align__(16) unsigned AsD[2][64 * SA];
  __shared__ __align__(16) unsigned BsD[2][64 * SA];

  const int tid = threadIdx.x;
  const int w = tid >> 5, l = tid & 31;
  const int lm = l & 15, half = l >> 4;
  const int rr = tid >> 2, c4 = tid & 3;          // stage slot 0 (rows 0..31)
  const int rr2 = rr + 32;                        // stage slot 1 (rows 32..63)

  CFr acc[4];
  #pragma unroll
  for (int i = 0; i < 4; ++i)
    #pragma unroll
    for (int r = 0; r < 8; ++r) acc[i].f[r] = 0.f;

  // stage(kt, buf): 64 rows x 64B per matrix via async b128 (16B per lane)
  #define STAGE(ktv, bufv)                                                          \
    do {                                                                            \
      async_ld16((unsigned)(uintptr_t)&AsD[bufv][rr * SA + c4 * 4],                 \
                 &xnQ[(size_t)(mBase + rr) * 64 + (ktv) * 4 + c4]);                 \
      async_ld16((unsigned)(uintptr_t)&BsD[bufv][rr * SA + c4 * 4],                 \
                 &wQ [(size_t)(nBase + rr) * 64 + (ktv) * 4 + c4]);                 \
      async_ld16((unsigned)(uintptr_t)&AsD[bufv][rr2 * SA + c4 * 4],                \
                 &xnQ[(size_t)(mBase + rr2) * 64 + (ktv) * 4 + c4]);                \
      async_ld16((unsigned)(uintptr_t)&BsD[bufv][rr2 * SA + c4 * 4],                \
                 &wQ [(size_t)(nBase + rr2) * 64 + (ktv) * 4 + c4]);                \
    } while (0)

  STAGE(0, 0);
  wait_async0();
  __syncthreads();

  int buf = 0;
  for (int kt = 0; kt < 16; ++kt) {
    if (kt < 15) STAGE(kt + 1, buf ^ 1);   // overlap next tile with compute
    ABf af;
    {
      uint4 q0 = *(const uint4*)&AsD[buf][(w * 16 + lm) * SA + 8 * half];
      uint4 q1 = *(const uint4*)&AsD[buf][(w * 16 + lm) * SA + 8 * half + 4];
      af.u[0] = q0.x; af.u[1] = q0.y; af.u[2] = q0.z; af.u[3] = q0.w;
      af.u[4] = q1.x; af.u[5] = q1.y; af.u[6] = q1.z; af.u[7] = q1.w;
    }
    #pragma unroll
    for (int nt = 0; nt < 4; ++nt) {
      ABf bfr;
      uint4 q0 = *(const uint4*)&BsD[buf][(nt * 16 + lm) * SA + 8 * half];
      uint4 q1 = *(const uint4*)&BsD[buf][(nt * 16 + lm) * SA + 8 * half + 4];
      bfr.u[0] = q0.x; bfr.u[1] = q0.y; bfr.u[2] = q0.z; bfr.u[3] = q0.w;
      bfr.u[4] = q1.x; bfr.u[5] = q1.y; bfr.u[6] = q1.z; bfr.u[7] = q1.w;
      acc[nt].v = __builtin_amdgcn_wmma_f32_16x16x32_bf16(
          false, af.v, false, bfr.v, (short)0, acc[nt].v, false, false);
    }
    if (kt < 15) {
      wait_async0();       // next buffer resident
      __syncthreads();     // all waves done reading current buffer
      buf ^= 1;
    }
  }
  #undef STAGE

  #pragma unroll
  for (int nt = 0; nt < 4; ++nt) {
    int n_g = nBase + nt * 16 + lm;
    float bias = bih[n_g];
    #pragma unroll
    for (int r = 0; r < 8; ++r) {
      int m_g = mBase + w * 16 + half * 8 + r;
      __builtin_nontemporal_store(acc[nt].f[r] + bias, &xpD[(size_t)m_g * G3 + n_g]);
    }
  }
}

// ---------------------------------------------------------------------------
// 4) Persistent bidirectional GRU recurrence + fused GELU output.
//    grid = 64 WGs x 64 thr. dir = blk>>5, 32 WGs per dir, 16 h-cols each.
//    Per step: WMMA [32x512]x[512x48] -> gh LDS -> gates -> h dbl-buffer +
//    GELU to out. Device-scope atomic barrier + full-block acquire/release
//    fences make plain b128 h loads/stores coherent across WGs.
// ---------------------------------------------------------------------------
__global__ __launch_bounds__(64) void gru_kernel(const float* __restrict__ xp,
                                                 const __bf16* __restrict__ whh,
                                                 const float* __restrict__ bhhf,
                                                 const float* __restrict__ bhhb,
                                                 unsigned* h16,           // bf16 pairs as dwords
                                                 float* h32,
                                                 unsigned* bar,
                                                 float* __restrict__ out) {
  const int dir   = blockIdx.x >> 5;
  const int wg    = blockIdx.x & 31;
  const int jbase = wg * 16;

  __shared__ __align__(16) unsigned wldsD[48 * SW]; // 3 gates x 16 rows x 512 bf16
  __shared__ float gh[32 * 52];                     // 32 batch x 48 gate-cols, padded

  const int tid = threadIdx.x;
  const uint4* whhQ = (const uint4*)whh + (size_t)dir * ((size_t)G3 * I_DIM / 8);
  for (int i = tid; i < 48 * 64; i += 64) {          // 48 rows x 64 quads
    int lr = i >> 6, c4 = i & 63;
    int g = lr >> 4, jj = lr & 15;
    int grow = g * H_DIM + jbase + jj;
    *(uint4*)&wldsD[lr * SW + c4 * 4] = whhQ[(size_t)grow * 64 + c4];
  }
  const float* bhh = dir ? bhhb : bhhf;
  const float* xpd = xp + (size_t)dir * (size_t)M_ROWS * G3;
  __syncthreads();

  const int w = tid >> 5, l = tid & 31;
  const int lm = l & 15, half = l >> 4;
  unsigned target = 0;

  for (int s = 0; s < T_DIM; ++s) {
    const int t = dir ? (T_DIM - 1 - s) : s;
    const int p = s & 1;
    const uint4* hprevQ = (const uint4*)(h16 + (dir * 2 + p) * 8192);

    // prefetch next step's xp slice (3 cachelines per batch row) during GEMM
    if (s + 1 < T_DIM && tid < 32) {
      int tn = dir ? (T_DIM - 2 - s) : (s + 1);
      const float* xpr = xpd + ((size_t)tn * 32 + tid) * G3 + jbase;
      __builtin_prefetch(xpr, 0, 1);
      __builtin_prefetch(xpr + H_DIM, 0, 1);
      __builtin_prefetch(xpr + 2 * H_DIM, 0, 1);
    }

    CFr acc[3];
    #pragma unroll
    for (int g = 0; g < 3; ++g)
      #pragma unroll
      for (int r = 0; r < 8; ++r) acc[g].f[r] = 0.f;

    const int m_g = w * 16 + lm;  // batch row for this lane
    for (int kt = 0; kt < 16; ++kt) {
      ABf af;
      {
        uint4 q0 = hprevQ[m_g * 64 + kt * 4 + 2 * half];       // dwords 8*half + 0..3
        uint4 q1 = hprevQ[m_g * 64 + kt * 4 + 2 * half + 1];   // dwords 8*half + 4..7
        af.u[0] = q0.x; af.u[1] = q0.y; af.u[2] = q0.z; af.u[3] = q0.w;
        af.u[4] = q1.x; af.u[5] = q1.y; af.u[6] = q1.z; af.u[7] = q1.w;
      }
      #pragma unroll
      for (int g = 0; g < 3; ++g) {
        ABf bfr;
        int lr = g * 16 + lm;
        uint4 q0 = *(const uint4*)&wldsD[lr * SW + kt * 16 + 8 * half];
        uint4 q1 = *(const uint4*)&wldsD[lr * SW + kt * 16 + 8 * half + 4];
        bfr.u[0] = q0.x; bfr.u[1] = q0.y; bfr.u[2] = q0.z; bfr.u[3] = q0.w;
        bfr.u[4] = q1.x; bfr.u[5] = q1.y; bfr.u[6] = q1.z; bfr.u[7] = q1.w;
        acc[g].v = __builtin_amdgcn_wmma_f32_16x16x32_bf16(
            false, af.v, false, bfr.v, (short)0, acc[g].v, false, false);
      }
    }
    #pragma unroll
    for (int g = 0; g < 3; ++g)
      #pragma unroll
      for (int r = 0; r < 8; ++r) {
        int m = w * 16 + half * 8 + r;
        gh[m * 52 + g * 16 + lm] = acc[g].f[r];
      }
    __syncthreads();

    float*    hprev32 = h32 + (dir * 2 + p) * 16384;
    float*    hnext32 = h32 + (dir * 2 + (p ^ 1)) * 16384;
    unsigned* hnextD  = h16 + (dir * 2 + (p ^ 1)) * 8192;

    #pragma unroll
    for (int k = 0; k < 4; ++k) {
      int idx2 = tid + k * 64;    // pair index 0..255
      int b = idx2 >> 3;          // 0..31
      int jp = idx2 & 7;          // 0..7  (pair of columns)
      unsigned pk = 0;
      #pragma unroll
      for (int e = 0; e < 2; ++e) {
        int jj = jp * 2 + e;
        int j = jbase + jj;
        float hr = gh[b * 52 + jj]      + bhh[j];
        float hz = gh[b * 52 + 16 + jj] + bhh[H_DIM + j];
        float hv = gh[b * 52 + 32 + jj] + bhh[2 * H_DIM + j];
        const float* xpr = xpd + ((size_t)t * 32 + b) * G3;
        float xr = xpr[j], xz = xpr[H_DIM + j], xnv = xpr[2 * H_DIM + j];
        float rr = 1.f / (1.f + expf(-(xr + hr)));
        float zz = 1.f / (1.f + expf(-(xz + hz)));
        float nn = tanhf(xnv + rr * hv);
        float hp = hprev32[b * H_DIM + j];
        float hnew = (1.f - zz) * nn + zz * hp;
        hnext32[b * H_DIM + j] = hnew;
        float gl = 0.5f * hnew * (1.f + erff(hnew * 0.70710678118654752f));
        __builtin_nontemporal_store(gl, &out[((size_t)b * T_DIM + t) * (2 * H_DIM) + dir * H_DIM + j]);
        BfPun pun; pun.h = (__bf16)hnew;
        pk |= ((unsigned)pun.u) << (16 * e);
      }
      hnextD[b * 256 + (jbase + jp * 2) / 2] = pk;
    }
    // release: make h writes visible at device scope (all threads fence)
    __threadfence();
    __syncthreads();
    target += 32;
    if (tid == 0) {
      atomicAdd(&bar[dir], 1u);
      while (__hip_atomic_load(&bar[dir], __ATOMIC_RELAXED, __HIP_MEMORY_SCOPE_AGENT) < target) {
        __builtin_amdgcn_s_sleep(1);
      }
    }
    __syncthreads();
    // acquire: every thread invalidates its CU's L0 so plain loads see new h
    __threadfence();
  }
}

// ---------------------------------------------------------------------------
extern "C" void kernel_launch(void* const* d_in, const int* in_sizes, int n_in,
                              void* d_out, int out_size, void* d_ws, size_t ws_size,
                              hipStream_t stream) {
  const float* x    = (const float*)d_in[0];
  const float* gam  = (const float*)d_in[1];
  const float* bet  = (const float*)d_in[2];
  const float* wihf = (const float*)d_in[3];
  const float* whhf = (const float*)d_in[4];
  const float* bihf = (const float*)d_in[5];
  const float* bhhf = (const float*)d_in[6];
  const float* wihb = (const float*)d_in[7];
  const float* whhb = (const float*)d_in[8];
  const float* bihb = (const float*)d_in[9];
  const float* bhhb = (const float*)d_in[10];
  float* out = (float*)d_out;

  char* ws = (char*)d_ws;
  size_t off = 0;
  __bf16* xn    = (__bf16*)(ws + off); off += (size_t)M_ROWS * I_DIM * 2;       // 32 MB
  __bf16* wih16 = (__bf16*)(ws + off); off += (size_t)2 * G3 * I_DIM * 2;       // 3 MB
  __bf16* whh16 = (__bf16*)(ws + off); off += (size_t)2 * G3 * I_DIM * 2;       // 3 MB
  float*  xp    = (float*) (ws + off); off += (size_t)2 * M_ROWS * G3 * 4;      // 384 MB
  unsigned* h16 = (unsigned*)(ws + off); size_t zbase = off; off += 2 * 2 * B_DIM * H_DIM * 2;
  float*  h32   = (float*) (ws + off); off += 2 * 2 * B_DIM * H_DIM * 4;
  unsigned* bar = (unsigned*)(ws + off); off += 256;

  // zero h double-buffers + barrier counters (contiguous region)
  hipMemsetAsync(ws + zbase, 0, off - zbase, stream);

  hipLaunchKernelGGL(ln_kernel,  dim3(M_ROWS), dim3(256), 0, stream, x, gam, bet, xn);
  hipLaunchKernelGGL(cvt_kernel, dim3(3072),   dim3(256), 0, stream,
                     wihf, wihb, whhf, whhb, wih16, whh16);
  hipLaunchKernelGGL(gemm_xp, dim3(G3 / 64, M_ROWS / 64, 2), dim3(128), 0, stream,
                     xn, wih16, bihf, bihb, xp);
  hipLaunchKernelGGL(gru_kernel, dim3(64), dim3(64), 0, stream,
                     xp, whh16, bhhf, bhhb, h16, h32, bar, out);
}
